// NOVA10M_16690242912265
// MI455X (gfx1250) — 8-bit matrix operand order repaired and checked by reading, compiled, not run on hardware
//
#include <hip/hip_runtime.h>
#include <hip/hip_bf16.h>
#include <stdint.h>

#define HID   640
#define DINN  1280
#define DCONV 4
#define DTR   40
#define NST   16
#define NEXP  5
#define RNK   8
#define NB    4
#define SEQ   1024
#define BL    4096
#define SCAL  2.0f

typedef int   v8i __attribute__((ext_vector_type(8)));
typedef float v8f __attribute__((ext_vector_type(8)));
typedef float v2f __attribute__((ext_vector_type(2)));

__global__ __launch_bounds__(256) void k_walpha(const float* __restrict__ w, int n,
                                                float* __restrict__ alpha) {
    __shared__ float red[256];
    float s = 0.f;
    for (int i = threadIdx.x; i < n; i += 256) s += fabsf(w[i]);
    red[threadIdx.x] = s; __syncthreads();
    for (int st = 128; st > 0; st >>= 1) {
        if (threadIdx.x < st) red[threadIdx.x] += red[threadIdx.x + st];
        __syncthreads();
    }
    if (threadIdx.x == 0) *alpha = fmaxf(red[0] / (float)n, 1e-10f);
}

__global__ __launch_bounds__(256) void k_wquant(const float* __restrict__ W,
                                                const float* __restrict__ alpha,
                                                int8_t* __restrict__ out,
                                                int O, int K, int Npad) {
    int i = blockIdx.x * 256 + threadIdx.x;
    if (i >= Npad * K) return;
    int n = i / K, k = i - n * K;
    int8_t q = 0;
    if (n < O) {
        float v = rintf(W[(size_t)n * K + k] / (*alpha));
        v = fminf(fmaxf(v, -1.f), 1.f);
        q = (int8_t)v;
    }
    out[i] = q;
}

__global__ __launch_bounds__(256) void k_rmsnorm(const float* __restrict__ x,
                                                 const float* __restrict__ w,
                                                 float* __restrict__ out, int I) {
    int m = blockIdx.x, tid = threadIdx.x;
    const float* row = x + (size_t)m * I;
    __shared__ float red[256];
    __shared__ float s_inv;
    float ss = 0.f;
    for (int i = tid; i < I; i += 256) { float v = row[i]; ss += v * v; }
    red[tid] = ss; __syncthreads();
    for (int st = 128; st > 0; st >>= 1) {
        if (tid < st) red[tid] += red[tid + st];
        __syncthreads();
    }
    if (tid == 0) s_inv = rsqrtf(red[0] / (float)I + 1e-6f);
    __syncthreads();
    float inv = s_inv;
    for (int i = tid; i < I; i += 256) out[(size_t)m * I + i] = row[i] * inv * w[i];
}

__global__ __launch_bounds__(256) void k_actquant(const float* __restrict__ x,
                                                  const float* __restrict__ nw,
                                                  int8_t* __restrict__ q,
                                                  float* __restrict__ gamma, int I) {
    int m = blockIdx.x, tid = threadIdx.x;
    const float* row = x + (size_t)m * I;
    __shared__ float red[256];
    __shared__ float s_inv, s_g;
    float ss = 0.f;
    for (int i = tid; i < I; i += 256) { float v = row[i]; ss += v * v; }
    red[tid] = ss; __syncthreads();
    for (int st = 128; st > 0; st >>= 1) {
        if (tid < st) red[tid] += red[tid + st];
        __syncthreads();
    }
    if (tid == 0) s_inv = rsqrtf(red[0] / (float)I + 1e-6f);
    __syncthreads();
    float inv = s_inv, mx = 0.f;
    for (int i = tid; i < I; i += 256) mx = fmaxf(mx, fabsf(row[i] * inv * nw[i]));
    red[tid] = mx; __syncthreads();
    for (int st = 128; st > 0; st >>= 1) {
        if (tid < st) red[tid] = fmaxf(red[tid], red[tid + st]);
        __syncthreads();
    }
    if (tid == 0) { s_g = fmaxf(red[0], 1e-10f); gamma[m] = s_g; }
    __syncthreads();
    float sc = 127.0f / s_g;
    for (int i = tid; i < I; i += 256) {
        float v = rintf(row[i] * inv * nw[i] * sc);
        v = fminf(fmaxf(v, -128.f), 127.f);
        q[(size_t)m * I + i] = (int8_t)v;
    }
}

__global__ __launch_bounds__(256) void k_gemm_iu8(const int8_t* __restrict__ Aq,
                                                  const int8_t* __restrict__ Wq,
                                                  const float* __restrict__ gamma,
                                                  const float* __restrict__ alpha,
                                                  const float* __restrict__ res,
                                                  float* __restrict__ out,
                                                  int K, int Npad) {
    int lane = threadIdx.x & 31;
    int wave = threadIdx.x >> 5;
    int wm = wave >> 1, wn = wave & 1;
    int half = lane >> 4, l15 = lane & 15;
    int tileM = blockIdx.x * 64 + wm * 16;
    int tileN = blockIdx.y * 32 + wn * 16;
    const int8_t* arow = Aq + (size_t)(tileM + l15) * K;
    const int8_t* brow = Wq + (size_t)(tileN + l15) * K;
    v8i acc = {};
    for (int kb = 0; kb < K; kb += 64) {
        v8i a, b;
#pragma unroll
        for (int v = 0; v < 8; ++v) {
            int kA = kb + ((v >> 1) << 4) + (half << 3) + ((v & 1) << 2);
            a[v] = *(const int*)(arow + kA);
            int kB = kb + ((v >> 1) << 4) + (half << 3) + ((v & 1) << 2);
            b[v] = *(const int*)(brow + kB);
        }
        acc = __builtin_amdgcn_wmma_i32_16x16x64_iu8(true, a, true, b, acc, false, false);
    }
    float av = (*alpha) * (1.0f / 127.0f);
    int n = tileN + l15;
#pragma unroll
    for (int r = 0; r < 8; ++r) {
        int m = tileM + half * 8 + r;
        float val = (float)acc[r] * av * gamma[m];
        if (res) val += res[(size_t)m * Npad + n];
        out[(size_t)m * Npad + n] = val;
    }
}

__global__ __launch_bounds__(256) void k_gemm_f32(const float* __restrict__ A, int lda,
                                                  const float* __restrict__ W, int ldw,
                                                  const float* __restrict__ bias,
                                                  float* __restrict__ out,
                                                  int N, int K, int act) {
    int lane = threadIdx.x & 31;
    int wave = threadIdx.x >> 5;
    int wm = wave >> 1, wn = wave & 1;
    int half = lane >> 4, l15 = lane & 15;
    int tileM = blockIdx.x * 64 + wm * 16;
    int tileN = blockIdx.y * 32 + wn * 16;
    const float* arow = A + (size_t)(tileM + l15) * lda + half * 2;
    const float* brow = W + (size_t)(tileN + l15) * ldw + half * 2;
    v8f acc = {};
    for (int kb = 0; kb < K; kb += 4) {
        v2f a, b;
        a[0] = arow[kb]; a[1] = arow[kb + 1];
        b[0] = brow[kb]; b[1] = brow[kb + 1];
        acc = __builtin_amdgcn_wmma_f32_16x16x4_f32(false, a, false, b, (short)0, acc,
                                                    false, false);
    }
    int n = tileN + l15;
    float bv = bias ? bias[n] : 0.f;
#pragma unroll
    for (int r = 0; r < 8; ++r) {
        int m = tileM + half * 8 + r;
        float v = acc[r] + bv;
        if (act == 1) v = (v > 20.f) ? v : log1pf(__expf(v));
        else if (act == 2) v = v / (1.f + __expf(-v));
        out[(size_t)m * N + n] = v;
    }
}

__global__ __launch_bounds__(256) void k_conv(const float* __restrict__ xz,
                                              const float* __restrict__ cw,
                                              const float* __restrict__ cb,
                                              float* __restrict__ out) {
    int i = blockIdx.x * 256 + threadIdx.x;
    if (i >= BL * DINN) return;
    int m = i / DINN, d = i - m * DINN;
    int b = m >> 10, l = m & 1023;
    float acc = cb[d];
#pragma unroll
    for (int k = 0; k < DCONV; ++k) {
        int ll = l - k;
        if (ll >= 0) acc += xz[(size_t)((b << 10) + ll) * (2 * DINN) + d] * cw[d * DCONV + k];
    }
    out[i] = acc / (1.f + __expf(-acc));
}

__global__ __launch_bounds__(256) void k_scan(const float* __restrict__ dtp,
                                              const float* __restrict__ xc,
                                              const float* __restrict__ Bm,
                                              const float* __restrict__ Cm,
                                              const float* __restrict__ Alog,
                                              float* __restrict__ ys) {
    int ch = blockIdx.x * 256 + threadIdx.x;
    int b = ch / DINN, d = ch - b * DINN;
    float Anf[NST], h[NST];
#pragma unroll
    for (int n = 0; n < NST; ++n) { Anf[n] = -__expf(Alog[d * NST + n]); h[n] = 0.f; }
    __shared__ float sB[NST], sC[NST];
    for (int l = 0; l < SEQ; ++l) {
        size_t base = (size_t)(b * SEQ + l);
        if (threadIdx.x < NST) sB[threadIdx.x] = Bm[base * 32 + threadIdx.x];
        else if (threadIdx.x < 2 * NST) sC[threadIdx.x - NST] = Cm[base * 32 + threadIdx.x - NST];
        __syncthreads();
        float dtv = dtp[base * DINN + d];
        float xv  = xc[base * DINN + d];
        float y = 0.f;
#pragma unroll
        for (int n = 0; n < NST; ++n) {
            float dA = __expf(Anf[n] * dtv);
            h[n] = dA * h[n] + dtv * sB[n] * xv;
            y += h[n] * sC[n];
        }
        ys[base * DINN + d] = y;
        __syncthreads();
    }
}

__global__ __launch_bounds__(256) void k_gate(const float* __restrict__ ys,
                                              const float* __restrict__ xc,
                                              const float* __restrict__ Dp,
                                              const float* __restrict__ xz,
                                              float* __restrict__ y2) {
    int i = blockIdx.x * 256 + threadIdx.x;
    if (i >= BL * DINN) return;
    int m = i / DINN, d = i - m * DINN;
    float y = ys[i] + xc[i] * Dp[d];
    float z = xz[(size_t)m * (2 * DINN) + DINN + d];
    y2[i] = y * (z / (1.f + __expf(-z)));
}

__global__ __launch_bounds__(256) void k_molora(const float* __restrict__ H,
                                                const float* __restrict__ R1,
                                                const float* __restrict__ w2,
                                                const float* __restrict__ b2,
                                                const float* __restrict__ lA,
                                                const float* __restrict__ lB,
                                                float* __restrict__ out) {
    int m = blockIdx.x, tid = threadIdx.x;
    __shared__ float sh[HID];
    __shared__ float slog[NEXP];
    __shared__ float st[2 * RNK];
    __shared__ float sw[2];
    __shared__ int   si[2];
    for (int i = tid; i < HID; i += 256) sh[i] = H[(size_t)m * HID + i];
    __syncthreads();
    if (tid < NEXP) {
        const float* r1 = R1 + (size_t)m * 256;
        float s = b2[tid];
        for (int j = 0; j < 256; ++j) s += r1[j] * w2[tid * 256 + j];
        slog[tid] = s;
    }
    __syncthreads();
    if (tid == 0) {
        float mx = slog[0];
        for (int e = 1; e < NEXP; ++e) mx = fmaxf(mx, slog[e]);
        float ex[NEXP];
        for (int e = 0; e < NEXP; ++e) ex[e] = __expf(slog[e] - mx);
        int i0 = 0;
        for (int e = 1; e < NEXP; ++e) if (ex[e] > ex[i0]) i0 = e;
        int i1 = (i0 == 0) ? 1 : 0;
        for (int e = 0; e < NEXP; ++e) if (e != i0 && ex[e] > ex[i1]) i1 = e;
        float den = ex[i0] + ex[i1];
        sw[0] = ex[i0] / den; sw[1] = ex[i1] / den;
        si[0] = i0; si[1] = i1;
    }
    __syncthreads();
    if (tid < 2 * RNK) {
        int slot = tid >> 3, r = tid & 7, e = si[slot];
        float s = 0.f;
        for (int hh = 0; hh < HID; ++hh) s += sh[hh] * lA[((size_t)e * HID + hh) * RNK + r];
        st[tid] = s;
    }
    __syncthreads();
    for (int hh = tid; hh < HID; hh += 256) {
        float acc = sh[hh];
#pragma unroll
        for (int slot = 0; slot < 2; ++slot) {
            int e = si[slot];
            float s = 0.f;
#pragma unroll
            for (int r = 0; r < RNK; ++r) s += st[slot * RNK + r] * lB[((size_t)e * RNK + r) * HID + hh];
            acc += sw[slot] * SCAL * s;
        }
        out[(size_t)m * HID + hh] = acc;
    }
}

extern "C" void kernel_launch(void* const* d_in, const int* in_sizes, int n_in,
                              void* d_out, int out_size, void* d_ws, size_t ws_size,
                              hipStream_t stream) {
    const float* x     = (const float*)d_in[0];
    const float* mnw   = (const float*)d_in[1];
    const float* inW   = (const float*)d_in[2];
    const float* inNW  = (const float*)d_in[3];
    const float* convW = (const float*)d_in[4];
    const float* convB = (const float*)d_in[5];
    const float* dtW   = (const float*)d_in[6];
    const float* dtNW  = (const float*)d_in[7];
    const float* dtDW  = (const float*)d_in[8];
    const float* dtDB  = (const float*)d_in[9];
    const float* BW    = (const float*)d_in[10];
    const float* BNW   = (const float*)d_in[11];
    const float* CW    = (const float*)d_in[12];
    const float* CNW   = (const float*)d_in[13];
    const float* Alog  = (const float*)d_in[14];
    const float* Dp    = (const float*)d_in[15];
    const float* outW  = (const float*)d_in[16];
    const float* outNW = (const float*)d_in[17];
    const float* r1W   = (const float*)d_in[18];
    const float* r1B   = (const float*)d_in[19];
    const float* r2W   = (const float*)d_in[20];
    const float* r2B   = (const float*)d_in[21];
    const float* lA    = (const float*)d_in[22];
    const float* lB    = (const float*)d_in[23];
    float* outp = (float*)d_out;

    char* ws = (char*)d_ws;
    size_t o = 0;
    auto take = [&](size_t bytes) { size_t r = o; o = (o + bytes + 255) & ~(size_t)255; return r; };
    float*  ALPHA = (float*)(ws + take(NEXP * 4));
    int8_t* WQIN  = (int8_t*)(ws + take((size_t)2 * DINN * HID));
    int8_t* WQDT  = (int8_t*)(ws + take((size_t)64 * DINN));
    int8_t* WQB   = (int8_t*)(ws + take((size_t)32 * DINN));
    int8_t* WQC   = (int8_t*)(ws + take((size_t)32 * DINN));
    int8_t* WQO   = (int8_t*)(ws + take((size_t)HID * DINN));
    float*  H1    = (float*)(ws + take((size_t)BL * HID * 4));
    int8_t* XQ    = (int8_t*)(ws + take((size_t)BL * DINN));
    float*  GAM   = (float*)(ws + take((size_t)BL * 4));
    float*  XZ    = (float*)(ws + take((size_t)BL * 2 * DINN * 4));
    float*  XC    = (float*)(ws + take((size_t)BL * DINN * 4));
    float*  DTUP  = (float*)(ws + take((size_t)BL * 64 * 4));
    float*  DT    = (float*)(ws + take((size_t)BL * DINN * 4));
    float*  BM    = (float*)(ws + take((size_t)BL * 32 * 4));
    float*  CM    = (float*)(ws + take((size_t)BL * 32 * 4));
    float*  YS    = (float*)(ws + take((size_t)BL * DINN * 4));
    float*  Y2    = (float*)(ws + take((size_t)BL * DINN * 4));
    float*  HOUT  = (float*)(ws + take((size_t)BL * HID * 4));
    float*  R1    = (float*)(ws + take((size_t)BL * 256 * 4));
    (void)ws_size; (void)in_sizes; (void)n_in; (void)out_size;

    dim3 b256(256);
    hipLaunchKernelGGL(k_walpha, dim3(1), b256, 0, stream, inW, 2 * DINN * HID, ALPHA + 0);
    hipLaunchKernelGGL(k_walpha, dim3(1), b256, 0, stream, dtW, DTR * DINN, ALPHA + 1);
    hipLaunchKernelGGL(k_walpha, dim3(1), b256, 0, stream, BW, NST * DINN, ALPHA + 2);
    hipLaunchKernelGGL(k_walpha, dim3(1), b256, 0, stream, CW, NST * DINN, ALPHA + 3);
    hipLaunchKernelGGL(k_walpha, dim3(1), b256, 0, stream, outW, HID * DINN, ALPHA + 4);
    hipLaunchKernelGGL(k_wquant, dim3((2 * DINN * HID + 255) / 256), b256, 0, stream,
                       inW, ALPHA + 0, WQIN, 2 * DINN, HID, 2 * DINN);
    hipLaunchKernelGGL(k_wquant, dim3((64 * DINN + 255) / 256), b256, 0, stream,
                       dtW, ALPHA + 1, WQDT, DTR, DINN, 64);
    hipLaunchKernelGGL(k_wquant, dim3((32 * DINN + 255) / 256), b256, 0, stream,
                       BW, ALPHA + 2, WQB, NST, DINN, 32);
    hipLaunchKernelGGL(k_wquant, dim3((32 * DINN + 255) / 256), b256, 0, stream,
                       CW, ALPHA + 3, WQC, NST, DINN, 32);
    hipLaunchKernelGGL(k_wquant, dim3((HID * DINN + 255) / 256), b256, 0, stream,
                       outW, ALPHA + 4, WQO, HID, DINN, 64 * 10);

    hipLaunchKernelGGL(k_rmsnorm, dim3(BL), b256, 0, stream, x, mnw, H1, HID);
    hipLaunchKernelGGL(k_actquant, dim3(BL), b256, 0, stream, H1, inNW, XQ, GAM, HID);
    hipLaunchKernelGGL(k_gemm_iu8, dim3(BL / 64, (2 * DINN) / 32), b256, 0, stream,
                       XQ, WQIN, GAM, ALPHA + 0, (const float*)nullptr, XZ, HID, 2 * DINN);

    hipLaunchKernelGGL(k_conv, dim3((BL * DINN + 255) / 256), b256, 0, stream,
                       XZ, convW, convB, XC);

    hipLaunchKernelGGL(k_actquant, dim3(BL), b256, 0, stream, XC, dtNW, XQ, GAM, DINN);
    hipLaunchKernelGGL(k_gemm_iu8, dim3(BL / 64, 64 / 32), b256, 0, stream,
                       XQ, WQDT, GAM, ALPHA + 1, (const float*)nullptr, DTUP, DINN, 64);
    hipLaunchKernelGGL(k_actquant, dim3(BL), b256, 0, stream, XC, BNW, XQ, GAM, DINN);
    hipLaunchKernelGGL(k_gemm_iu8, dim3(BL / 64, 1), b256, 0, stream,
                       XQ, WQB, GAM, ALPHA + 2, (const float*)nullptr, BM, DINN, 32);
    hipLaunchKernelGGL(k_actquant, dim3(BL), b256, 0, stream, XC, CNW, XQ, GAM, DINN);
    hipLaunchKernelGGL(k_gemm_iu8, dim3(BL / 64, 1), b256, 0, stream,
                       XQ, WQC, GAM, ALPHA + 3, (const float*)nullptr, CM, DINN, 32);

    hipLaunchKernelGGL(k_gemm_f32, dim3(BL / 64, DINN / 32), b256, 0, stream,
                       DTUP, 64, dtDW, DTR, dtDB, DT, DINN, DTR, 1);

    hipLaunchKernelGGL(k_scan, dim3(NB * DINN / 256), b256, 0, stream,
                       DT, XC, BM, CM, Alog, YS);

    hipLaunchKernelGGL(k_gate, dim3((BL * DINN + 255) / 256), b256, 0, stream,
                       YS, XC, Dp, XZ, Y2);

    hipLaunchKernelGGL(k_actquant, dim3(BL), b256, 0, stream, Y2, outNW, XQ, GAM, DINN);
    hipLaunchKernelGGL(k_gemm_iu8, dim3(BL / 64, HID / 32), b256, 0, stream,
                       XQ, WQO, GAM, ALPHA + 4, x, HOUT, DINN, HID);

    hipLaunchKernelGGL(k_gemm_f32, dim3(BL / 64, 256 / 32), b256, 0, stream,
                       HOUT, HID, r1W, HID, r1B, R1, 256, HID, 2);
    hipLaunchKernelGGL(k_molora, dim3(BL), b256, 0, stream,
                       HOUT, R1, r2W, r2B, lA, lB, outp);
}
